// attn1d_56272661512557
// MI455X (gfx1250) — compile-verified, matrix operand order checked by reading, not run on hardware
//
#include <hip/hip_runtime.h>

// ---------------------------------------------------------------------------
// Types for CDNA5 WMMA (wave32): V_WMMA_F32_16X16X32_BF16
// ---------------------------------------------------------------------------
typedef __bf16 bf16_t;
typedef __attribute__((ext_vector_type(8)))  __bf16 v8bf;
typedef __attribute__((ext_vector_type(16))) __bf16 v16bf;
typedef __attribute__((ext_vector_type(4)))  float  v4f;
typedef __attribute__((ext_vector_type(8)))  float  v8f;

// Native f32 -> bf16 convert (RNE); lowers to v_cvt_pk_bf16_f32 pairs.
static __device__ __forceinline__ bf16_t f2bf(float f) { return (bf16_t)f; }

static __device__ __forceinline__ v8bf pack8(v4f a, v4f b) {
    v8bf r;
    r[0] = f2bf(a[0]); r[1] = f2bf(a[1]); r[2] = f2bf(a[2]); r[3] = f2bf(a[3]);
    r[4] = f2bf(b[0]); r[5] = f2bf(b[1]); r[6] = f2bf(b[2]); r[7] = f2bf(b[3]);
    return r;
}

static __device__ __forceinline__ v8f wmma_bf16(v16bf a, v16bf b, v8f c) {
    // (neg_a, A, neg_b, B, c_mod, C, reuse_a, reuse_b)
    return __builtin_amdgcn_wmma_f32_16x16x32_bf16(false, a, false, b, (short)0, c,
                                                   false, false);
}

static __device__ __forceinline__ v16bf combine8(v8bf lo, v8bf hi) {
    return __builtin_shufflevector(lo, hi, 0, 1, 2, 3, 4, 5, 6, 7,
                                           8, 9, 10, 11, 12, 13, 14, 15);
}

// A/B fragment from an LDS row laid out [.., 32 k] bf16 (16B-aligned chunks):
// elements 0..7 from p[0..7], 8..15 from p[16..23] per ISA K-interleave.
static __device__ __forceinline__ v16bf load_frag_lds(const bf16_t* p) {
    return combine8(*(const v8bf*)p, *(const v8bf*)(p + 16));
}

// Same fragment pattern sourced from global f32 with on-the-fly bf16 cvt.
static __device__ __forceinline__ v16bf load_frag_f32(const float* p) {
    v4f f0 = *(const v4f*)(p + 0);
    v4f f1 = *(const v4f*)(p + 4);
    v4f f2 = *(const v4f*)(p + 16);
    v4f f3 = *(const v4f*)(p + 20);
    return combine8(pack8(f0, f1), pack8(f2, f3));
}

#define V8F_ZERO {0.f,0.f,0.f,0.f,0.f,0.f,0.f,0.f}

// ---------------------------------------------------------------------------
// Generic bf16-WMMA GEMM: C = epilogue(alpha * (A @ W) + bias [+relu] [+res])
// A: [M,K] f32 row-major, W: [K,N] f32 row-major. M % 256, N % 128, K % 32.
// WG: 256 thr (8 waves), WG tile 256x128, wave tile 64x64 (16 WMMAs / 16
// ds_load_b128 per K-step), K-step 32, software-pipelined via reg preload.
// ---------------------------------------------------------------------------
#define BM 256
#define BN 128
#define TK 32

__global__ __launch_bounds__(256)
void gemm_bf16_kernel(const float* __restrict__ A, const float* __restrict__ W,
                      const float* __restrict__ bias, const float* __restrict__ residual,
                      float* __restrict__ C, int M, int N, int K,
                      float alpha, int do_relu)
{
    __shared__ __attribute__((aligned(16))) bf16_t At[BM * TK];   // [m][k]  16 KB
    __shared__ __attribute__((aligned(16))) bf16_t Bt[BN * TK];   // [n][k]   8 KB

    const int tid  = threadIdx.x;
    const int lane = tid & 31;
    const int wave = tid >> 5;
    const int li   = lane & 15;
    const int half = lane >> 4;
    const int k0   = half * 8;
    const int wm   = wave & 3;    // 4 row groups x 64 rows
    const int wn   = wave >> 2;   // 2 col groups x 64 cols

    const int m0 = blockIdx.y * BM;
    const int n0 = blockIdx.x * BN;

    v8f acc[4][4];
#pragma unroll
    for (int mt = 0; mt < 4; ++mt)
#pragma unroll
        for (int nt = 0; nt < 4; ++nt) { v8f z = V8F_ZERO; acc[mt][nt] = z; }

    // A loader: one 32-wide k row per thread (256 rows / 256 threads).
    // B loader: 8 threads per k-row, 16 n each.
    const int b_kk  = tid >> 3;             // 0..31
    const int b_seg = (tid & 7) * 16;       // 0..112

    const float* aptr = A + (size_t)(m0 + tid) * K;             // + kb
    const float* bptr = W + (size_t)b_kk * N + n0 + b_seg;      // + kb*N

    // Prologue: preload tile kb=0 into registers (A: 8 x v4f, B: 4 x v4f).
    v4f ar[8], br[4];
#pragma unroll
    for (int c = 0; c < 8; ++c) ar[c] = *(const v4f*)(aptr + c * 4);
#pragma unroll
    for (int c = 0; c < 4; ++c) br[c] = *(const v4f*)(bptr + c * 4);

    for (int kb = 0; kb < K; kb += TK) {
        // Convert staged registers -> LDS (A: 4x ds_store_b128, contiguous).
        {
            v8bf* dst = (v8bf*)&At[tid * TK];
            dst[0] = pack8(ar[0], ar[1]);
            dst[1] = pack8(ar[2], ar[3]);
            dst[2] = pack8(ar[4], ar[5]);
            dst[3] = pack8(ar[6], ar[7]);
        }
#pragma unroll
        for (int c = 0; c < 4; ++c) {       // transposed W tile: [n][k]
            const int nn = b_seg + c * 4;
            Bt[(nn + 0) * TK + b_kk] = f2bf(br[c][0]);
            Bt[(nn + 1) * TK + b_kk] = f2bf(br[c][1]);
            Bt[(nn + 2) * TK + b_kk] = f2bf(br[c][2]);
            Bt[(nn + 3) * TK + b_kk] = f2bf(br[c][3]);
        }
        __syncthreads();

        // Issue next tile's global loads; they overlap the WMMAs below.
        if (kb + TK < K) {
            const float* an = aptr + kb + TK;
            const float* bn = bptr + (size_t)(kb + TK) * N;
#pragma unroll
            for (int c = 0; c < 8; ++c) ar[c] = *(const v4f*)(an + c * 4);
#pragma unroll
            for (int c = 0; c < 4; ++c) br[c] = *(const v4f*)(bn + c * 4);
        }

        // 4 A-frags + 4 B-frags -> 16 WMMAs (1.0 ds_load_b128 per WMMA).
        v16bf bfrag[4];
#pragma unroll
        for (int nt = 0; nt < 4; ++nt)
            bfrag[nt] = load_frag_lds(&Bt[(wn * 64 + nt * 16 + li) * TK + k0]);
#pragma unroll
        for (int mt = 0; mt < 4; ++mt) {
            const v16bf afrag =
                load_frag_lds(&At[(wm * 64 + mt * 16 + li) * TK + k0]);
#pragma unroll
            for (int nt = 0; nt < 4; ++nt)
                acc[mt][nt] = wmma_bf16(afrag, bfrag[nt], acc[mt][nt]);
        }
        __syncthreads();
    }

    // Epilogue. C/D layout: lanes 0-15 rows M=0..7 (VGPR r), lanes 16-31 rows 8..15.
#pragma unroll
    for (int mt = 0; mt < 4; ++mt) {
#pragma unroll
        for (int nt = 0; nt < 4; ++nt) {
            const int col = n0 + wn * 64 + nt * 16 + li;
            const float bv = bias ? bias[col] : 0.0f;
#pragma unroll
            for (int r = 0; r < 8; ++r) {
                const int row = m0 + wm * 64 + mt * 16 + half * 8 + r;
                float x = acc[mt][nt][r] * alpha + bv;
                if (do_relu) x = fmaxf(x, 0.0f);
                if (residual) x += residual[(size_t)row * N + col];
                C[(size_t)row * N + col] = x;
            }
        }
    }
}

// ---------------------------------------------------------------------------
// Flash attention per head. Heads split via pure reshape => Q/K/V head blocks
// are contiguous [1024 x 64] at (b*16+h)*65536; rel_h has row stride 1024,
// col offset h*64.  WG: 256 thr = 8 waves; wave owns 16 q rows; loop over
// 64-key blocks staged in LDS (K, Rel, V^T), software-pipelined.
// ---------------------------------------------------------------------------
#define DH 64
#define KBLK 64
#define QROWS 128

__global__ __launch_bounds__(256)
void flash_attn_kernel(const float* __restrict__ Q, const float* __restrict__ Km,
                       const float* __restrict__ Vm, const float* __restrict__ rel,
                       float* __restrict__ O, int N, int NHh, int DKtot)
{
    __shared__ __attribute__((aligned(16))) bf16_t Kt[KBLK * DH];     // [key][d]
    __shared__ __attribute__((aligned(16))) bf16_t Rt[KBLK * DH];     // [key][d]
    __shared__ __attribute__((aligned(16))) bf16_t Vt[DH * KBLK];     // [d][key]
    __shared__ __attribute__((aligned(16))) bf16_t Pst[8 * 16 * DH];  // per-wave P staging

    const int tid  = threadIdx.x;
    const int lane = tid & 31;
    const int wave = tid >> 5;
    const int li   = lane & 15;
    const int half = lane >> 4;
    const int k0   = half * 8;

    const int bh = blockIdx.y;                 // b*NH + h
    const int h  = bh & (NHh - 1);
    const size_t headBase = (size_t)bh * N * DH;

    const int q0 = blockIdx.x * QROWS + wave * 16;

    // Q fragments (16 rows x 64 d = 2 A-fragments of K=32), bf16-converted once.
    v16bf qa[2];
    {
        const float* qp = Q + headBase + (size_t)(q0 + li) * DH;
        qa[0] = load_frag_f32(qp + 0  + k0);
        qa[1] = load_frag_f32(qp + 32 + k0);
    }

    float mrow[8], lrow[8];
    v8f oacc[4];
#pragma unroll
    for (int r = 0; r < 8; ++r) { mrow[r] = -3.0e38f; lrow[r] = 0.0f; }
#pragma unroll
    for (int dt = 0; dt < 4; ++dt) { v8f z = V8F_ZERO; oacc[dt] = z; }

    const int trow = tid >> 2;            // 0..63 : key row for tile loads
    const int tseg = (tid & 3) * 16;      // 16-col segment

    const float* kbase = Km  + headBase + (size_t)trow * DH    + tseg;
    const float* vbase = Vm  + headBase + (size_t)trow * DH    + tseg;
    const float* rbase = rel + (size_t)trow * DKtot + h * DH   + tseg;

    // Prologue: preload key-block 0 into registers.
    v4f kr[4], rr[4], vr[4];
#pragma unroll
    for (int c = 0; c < 4; ++c) {
        kr[c] = *(const v4f*)(kbase + c * 4);
        rr[c] = *(const v4f*)(rbase + c * 4);
        vr[c] = *(const v4f*)(vbase + c * 4);
    }

    for (int kb = 0; kb < N; kb += KBLK) {
        // Stage K / Rel (row-major, vector stores) and V^T (transposed, scalar).
        *(v8bf*)&Kt[trow * DH + tseg]     = pack8(kr[0], kr[1]);
        *(v8bf*)&Kt[trow * DH + tseg + 8] = pack8(kr[2], kr[3]);
        *(v8bf*)&Rt[trow * DH + tseg]     = pack8(rr[0], rr[1]);
        *(v8bf*)&Rt[trow * DH + tseg + 8] = pack8(rr[2], rr[3]);
#pragma unroll
        for (int c = 0; c < 4; ++c) {
            const int cc = tseg + c * 4;
            Vt[(cc + 0) * KBLK + trow] = f2bf(vr[c][0]);
            Vt[(cc + 1) * KBLK + trow] = f2bf(vr[c][1]);
            Vt[(cc + 2) * KBLK + trow] = f2bf(vr[c][2]);
            Vt[(cc + 3) * KBLK + trow] = f2bf(vr[c][3]);
        }
        __syncthreads();

        // Issue next key-block's global loads; overlap with compute below.
        if (kb + KBLK < N) {
            const float* kn = kbase + (size_t)(kb + KBLK) * DH;
            const float* vn = vbase + (size_t)(kb + KBLK) * DH;
            const float* rn = rbase + (size_t)(kb + KBLK) * DKtot;
#pragma unroll
            for (int c = 0; c < 4; ++c) {
                kr[c] = *(const v4f*)(kn + c * 4);
                rr[c] = *(const v4f*)(rn + c * 4);
                vr[c] = *(const v4f*)(vn + c * 4);
            }
        }

        // S(16x64) = Q K^T + Q Rel^T   (4 col-tiles x 4 WMMAs)
        v8f sacc[4];
#pragma unroll
        for (int ct = 0; ct < 4; ++ct) {
            v8f s = V8F_ZERO;
            const bf16_t* kp = &Kt[(ct * 16 + li) * DH + k0];
            const bf16_t* rp = &Rt[(ct * 16 + li) * DH + k0];
            s = wmma_bf16(qa[0], load_frag_lds(kp), s);
            s = wmma_bf16(qa[1], load_frag_lds(kp + 32), s);
            s = wmma_bf16(qa[0], load_frag_lds(rp), s);
            s = wmma_bf16(qa[1], load_frag_lds(rp + 32), s);
            sacc[ct] = s;
        }

        // Online softmax per row; a tile row spans 16 lanes -> shfl_xor(8..1).
        bf16_t* pst = &Pst[wave * 16 * DH];
#pragma unroll
        for (int r = 0; r < 8; ++r) {
            float mx = fmaxf(fmaxf(sacc[0][r], sacc[1][r]),
                             fmaxf(sacc[2][r], sacc[3][r]));
#pragma unroll
            for (int off = 8; off >= 1; off >>= 1)
                mx = fmaxf(mx, __shfl_xor(mx, off, 32));
            const float mnew = fmaxf(mrow[r], mx);
            const float corr = __expf(mrow[r] - mnew);
            float rs = 0.0f;
#pragma unroll
            for (int ct = 0; ct < 4; ++ct) {
                const float p = __expf(sacc[ct][r] - mnew);
                rs += p;
                pst[(half * 8 + r) * DH + ct * 16 + li] = f2bf(p);
            }
#pragma unroll
            for (int off = 8; off >= 1; off >>= 1)
                rs += __shfl_xor(rs, off, 32);
            lrow[r] = lrow[r] * corr + rs;
            mrow[r] = mnew;
#pragma unroll
            for (int dt = 0; dt < 4; ++dt)
                oacc[dt][r] = oacc[dt][r] * corr;
        }
        // same-wave LDS write->read transpose of P; drain DS pipe explicitly
        asm volatile("s_wait_dscnt 0" ::: "memory");

        // O += P V : A = P (16x64), B = V^T staged as [d][key]
        v16bf pa0 = load_frag_lds(&pst[li * DH + k0]);
        v16bf pa1 = load_frag_lds(&pst[li * DH + 32 + k0]);
#pragma unroll
        for (int dt = 0; dt < 4; ++dt) {
            const bf16_t* vp = &Vt[(dt * 16 + li) * KBLK + k0];
            oacc[dt] = wmma_bf16(pa0, load_frag_lds(vp), oacc[dt]);
            oacc[dt] = wmma_bf16(pa1, load_frag_lds(vp + 32), oacc[dt]);
        }
        __syncthreads();
    }

    // Normalize and store (contiguous head block; global reshape is a no-op).
    float* op = O + headBase;
#pragma unroll
    for (int r = 0; r < 8; ++r) {
        const float inv = 1.0f / lrow[r];
        const int row = q0 + half * 8 + r;
#pragma unroll
        for (int dt = 0; dt < 4; ++dt)
            op[(size_t)row * DH + dt * 16 + li] = oacc[dt][r] * inv;
    }
}

// ---------------------------------------------------------------------------
// LayerNorm over rows of length D (=1024), one 256-thread WG per row.
// ---------------------------------------------------------------------------
__global__ __launch_bounds__(256)
void layernorm_kernel(const float* __restrict__ X, const float* __restrict__ g,
                      const float* __restrict__ b, float* __restrict__ Y, int D)
{
    __shared__ float red[32];
    const int row = blockIdx.x;
    const int tid = threadIdx.x;
    const int lane = tid & 31, wave = tid >> 5;
    const float* x = X + (size_t)row * D;

    float s = 0.0f, s2 = 0.0f;
    for (int i = tid; i < D; i += 256) { const float v = x[i]; s += v; s2 += v * v; }
#pragma unroll
    for (int off = 16; off >= 1; off >>= 1) {
        s  += __shfl_xor(s, off, 32);
        s2 += __shfl_xor(s2, off, 32);
    }
    if (lane == 0) { red[wave] = s; red[8 + wave] = s2; }
    __syncthreads();
    if (tid < 32) {
        float a  = (lane < 8) ? red[lane] : 0.0f;
        float a2 = (lane < 8) ? red[8 + lane] : 0.0f;
#pragma unroll
        for (int off = 4; off >= 1; off >>= 1) {
            a  += __shfl_xor(a, off, 32);
            a2 += __shfl_xor(a2, off, 32);
        }
        if (lane == 0) { red[16] = a; red[17] = a2; }
    }
    __syncthreads();
    const float mean = red[16] / (float)D;
    const float var  = red[17] / (float)D - mean * mean;
    const float rstd = rsqrtf(var + 1e-5f);
    float* y = Y + (size_t)row * D;
    for (int i = tid; i < D; i += 256)
        y[i] = (x[i] - mean) * rstd * g[i] + b[i];
}

// ---------------------------------------------------------------------------
extern "C" void kernel_launch(void* const* d_in, const int* in_sizes, int n_in,
                              void* d_out, int out_size, void* d_ws, size_t ws_size,
                              hipStream_t stream)
{
    (void)in_sizes; (void)n_in; (void)out_size; (void)ws_size;

    const float* qi   = (const float*)d_in[0];
    const float* ki   = (const float*)d_in[1];
    const float* vi   = (const float*)d_in[2];
    const float* Wq   = (const float*)d_in[3];
    const float* Wk   = (const float*)d_in[4];
    const float* Wv   = (const float*)d_in[5];
    const float* Wz   = (const float*)d_in[6];
    const float* rel  = (const float*)d_in[7];
    const float* ln1g = (const float*)d_in[8];
    const float* ln1b = (const float*)d_in[9];
    const float* W1   = (const float*)d_in[10];
    const float* b1   = (const float*)d_in[11];
    const float* W2   = (const float*)d_in[12];
    const float* b2   = (const float*)d_in[13];
    const float* ln2g = (const float*)d_in[14];
    const float* ln2b = (const float*)d_in[15];
    float* out = (float*)d_out;

    const int Bb = 8, Nn = 1024, Cc = 1024, NHn = 16;
    const int M = Bb * Nn;                       // 8192
    const size_t SZ = (size_t)M * Cc;            // 8M floats = 32 MB

    float* ws = (float*)d_ws;
    float* qb = ws;                 // q projection
    float* kb = ws + SZ;            // k projection
    float* vb = ws + 2 * SZ;        // v projection
    float* ab = ws + 3 * SZ;        // attention output
    float* yb = qb;                 // reuse: attn@Wz + residual
    float* o1 = kb;                 // reuse: LN1 output (FFN input + residual)
    float* hb = vb;                 // reuse: relu(out1@W1+b1)
    float* o2 = ab;                 // reuse: h@W2 + b2 + out1

    const dim3 blk(256);
    const dim3 gemmGrid(Cc / BN, M / BM);        // (8, 32)
    const dim3 attnGrid(Nn / QROWS, Bb * NHn);   // (8, 128)
    const float qscale = 0.125f;                 // DKH^-0.5, folded into q proj

    gemm_bf16_kernel<<<gemmGrid, blk, 0, stream>>>(qi, Wq, nullptr, nullptr, qb,
                                                   M, Cc, Cc, qscale, 0);
    gemm_bf16_kernel<<<gemmGrid, blk, 0, stream>>>(ki, Wk, nullptr, nullptr, kb,
                                                   M, Cc, Cc, 1.0f, 0);
    gemm_bf16_kernel<<<gemmGrid, blk, 0, stream>>>(vi, Wv, nullptr, nullptr, vb,
                                                   M, Cc, Cc, 1.0f, 0);

    flash_attn_kernel<<<attnGrid, blk, 0, stream>>>(qb, kb, vb, rel, ab,
                                                    Nn, NHn, Cc);

    gemm_bf16_kernel<<<gemmGrid, blk, 0, stream>>>(ab, Wz, nullptr, qi, yb,
                                                   M, Cc, Cc, 1.0f, 0);
    layernorm_kernel<<<dim3(M), blk, 0, stream>>>(yb, ln1g, ln1b, o1, Cc);

    gemm_bf16_kernel<<<gemmGrid, blk, 0, stream>>>(o1, W1, b1, nullptr, hb,
                                                   M, Cc, Cc, 1.0f, 1);
    gemm_bf16_kernel<<<gemmGrid, blk, 0, stream>>>(hb, W2, b2, o1, o2,
                                                   M, Cc, Cc, 1.0f, 0);
    layernorm_kernel<<<dim3(M), blk, 0, stream>>>(o2, ln2g, ln2b, out, Cc);
}
